// CrossAttention_66623532696046
// MI455X (gfx1250) — matrix operand order repaired and checked by reading, not compiled, not run on hardware
//
#include <hip/hip_runtime.h>

// ---------------------------------------------------------------------------
// CrossAttention for MI455X (gfx1250, wave32, WMMA).
//   out = softmax((x Wq^T)(k Wk^T)^T * 1/sqrt(Dh)) (v Wv^T) Wo^T + bo
// B=2, NQ=NK=2048, DIM=1024, H=16, Dh=64.
// Matmul FLOPs via v_wmma_f32_16x16x32_bf16; K-tiles staged by the Tensor
// Data Mover (TENSORcnt); output-proj A tiles staged via async global->LDS
// (ASYNCcnt).
// ---------------------------------------------------------------------------

typedef __attribute__((ext_vector_type(16))) __bf16    v16bf;
typedef __attribute__((ext_vector_type(8)))  __bf16    v8bf;
typedef __attribute__((ext_vector_type(8)))  float     v8f;
typedef __attribute__((ext_vector_type(4)))  unsigned  v4u;
typedef __attribute__((ext_vector_type(4)))  int       v4i;
typedef __attribute__((ext_vector_type(8)))  int       v8i;

union FragBF { v16bf v; v8bf h[2]; };

__device__ __forceinline__ unsigned short bf16_bits(float f) {
  unsigned u = __builtin_bit_cast(unsigned, f);
  u += 0x7FFFu + ((u >> 16) & 1u);           // round-to-nearest-even
  return (unsigned short)(u >> 16);
}
__device__ __forceinline__ __bf16 f32_to_bf16(float f) {
  unsigned short b = bf16_bits(f);
  return __builtin_bit_cast(__bf16, b);
}
__device__ __forceinline__ float bf16_to_f32(__bf16 v) {
  unsigned u = ((unsigned)__builtin_bit_cast(unsigned short, v)) << 16;
  return __builtin_bit_cast(float, u);
}
__device__ __forceinline__ unsigned pack_bf16x2(float a, float b) {
  return (unsigned)bf16_bits(a) | ((unsigned)bf16_bits(b) << 16);
}

__device__ __forceinline__ v8f v8f_zero() {
  v8f z;
#pragma unroll
  for (int i = 0; i < 8; ++i) z[i] = 0.0f;
  return z;
}

// A-fragment (16x32 bf16, M across lanes&15, K split per lane-half):
//   lane<16: K = {0..7, 16..23};  lane>=16: K = {8..15, 24..31}
__device__ __forceinline__ v16bf load_a_frag(const __bf16* base, int stride, int lane) {
  int row = lane & 15;
  int klo = (lane < 16) ? 0 : 8;
  FragBF f;
  f.h[0] = *(const v8bf*)(base + row * stride + klo);
  f.h[1] = *(const v8bf*)(base + row * stride + klo + 16);
  return f.v;
}

// B-fragment (32x16 bf16, N across lanes&15, K split per lane-half like A):
// base points at [n0][k0] of a row-major [n][k] tile (i.e. B^T storage).
__device__ __forceinline__ v16bf load_b_frag(const __bf16* base, int stride, int lane) {
  int col = lane & 15;
  int klo = (lane < 16) ? 0 : 8;
  FragBF f;
  f.h[0] = *(const v8bf*)(base + col * stride + klo);
  f.h[1] = *(const v8bf*)(base + col * stride + klo + 16);
  return f.v;
}

__device__ __forceinline__ v8f wmma_bf16(v16bf a, v16bf b, v8f c) {
  return __builtin_amdgcn_wmma_f32_16x16x32_bf16(false, a, false, b, (short)0, c,
                                                 false, false);
}

// Async global->LDS (B128): lds dword addr in one VGPR, 64-bit global in pair.
__device__ __forceinline__ void async_load_b128(const void* gptr, void* ldsptr) {
  unsigned lds = (unsigned)(size_t)ldsptr;
  unsigned long long ga = (unsigned long long)(size_t)gptr;
  asm volatile("global_load_async_to_lds_b128 %0, %1, off"
               :: "v"(lds), "v"(ga) : "memory");
}
__device__ __forceinline__ void wait_asynccnt0() {
  asm volatile("s_wait_asynccnt 0x0" ::: "memory");
}

// ---------------------------------------------------------------------------
// GEMM:  C[M][N] = A[M][K] * W[N][K]^T   (torch Linear: y = x @ W.T)
// Block tile 128x128, K-step 32 (= WMMA K). 256 threads = 8 waves, 2x4 wave
// grid, each wave computes a 64x32 sub-tile (4x2 WMMA tiles, 8 WMMA/K-step).
// A_F32: A is f32 in global, converted to bf16 during LDS staging.
//        (bf16 A path uses global_load_async_to_lds_b128 instead)
// OUT_F32: write f32 + bias, else write bf16 (no bias).
// ---------------------------------------------------------------------------
#define GLDA 40   // padded LDS row stride (bf16 elems): 80B rows, 16B aligned

template <bool A_F32, bool OUT_F32>
__global__ __launch_bounds__(256) void gemm_kernel(const void* __restrict__ Aptr,
                                                   const float* __restrict__ W,
                                                   void* __restrict__ Cout,
                                                   const float* __restrict__ bias,
                                                   int M, int N, int K) {
  __shared__ __bf16 As[128 * GLDA];
  __shared__ __bf16 Bs[128 * GLDA];

  const int tid  = threadIdx.x;
  const int lane = tid & 31;
  const int wv   = tid >> 5;          // 0..7
  const int wM   = wv >> 2;           // 0..1  (64 rows each)
  const int wN   = wv & 3;            // 0..3  (32 cols each)
  const int m0   = blockIdx.y * 128;
  const int n0   = blockIdx.x * 128;

  v8f acc[4][2];
#pragma unroll
  for (int mt = 0; mt < 4; ++mt)
#pragma unroll
    for (int nt = 0; nt < 2; ++nt) acc[mt][nt] = v8f_zero();

  for (int k0 = 0; k0 < K; k0 += 32) {
    __syncthreads();
    // ---- stage A tile (128 x 32) ----
    if (A_F32) {
      const float* A = (const float*)Aptr;
#pragma unroll
      for (int i = 0; i < 4; ++i) {
        int c   = tid + i * 256;          // 1024 chunks of 4 floats
        int row = c >> 3;
        int k4  = (c & 7) << 2;
        const float4 f = *(const float4*)(A + (size_t)(m0 + row) * K + k0 + k4);
        unsigned* dst  = (unsigned*)(As + row * GLDA + k4);
        dst[0] = pack_bf16x2(f.x, f.y);
        dst[1] = pack_bf16x2(f.z, f.w);
      }
    } else {
      // bf16 A: async DMA global -> LDS, no VGPR round-trip
      const __bf16* A = (const __bf16*)Aptr;
#pragma unroll
      for (int i = 0; i < 2; ++i) {
        int c   = tid + i * 256;          // 512 chunks of 8 bf16 (16B)
        int row = c >> 2;
        int k8  = (c & 3) << 3;
        async_load_b128(A + (size_t)(m0 + row) * K + k0 + k8,
                        As + row * GLDA + k8);
      }
    }
    // ---- stage B tile: rows are W rows n0..n0+127 (B^T layout) ----
#pragma unroll
    for (int i = 0; i < 4; ++i) {
      int c   = tid + i * 256;
      int row = c >> 3;
      int k4  = (c & 7) << 2;
      const float4 f = *(const float4*)(W + (size_t)(n0 + row) * K + k0 + k4);
      unsigned* dst  = (unsigned*)(Bs + row * GLDA + k4);
      dst[0] = pack_bf16x2(f.x, f.y);
      dst[1] = pack_bf16x2(f.z, f.w);
    }
    if (k0 + 32 < K) {  // hint next tiles toward cache (global_prefetch)
      __builtin_prefetch((const char*)W + ((size_t)(n0 + (tid >> 1)) * K + k0 + 32) * 4, 0, 0);
    }
    if (!A_F32) wait_asynccnt0();
    __syncthreads();

    // ---- compute: 8 WMMAs per wave per K-step ----
    v16bf afr[4], bfr[2];
#pragma unroll
    for (int mt = 0; mt < 4; ++mt)
      afr[mt] = load_a_frag(As + (wM * 64 + mt * 16) * GLDA, GLDA, lane);
#pragma unroll
    for (int nt = 0; nt < 2; ++nt)
      bfr[nt] = load_b_frag(Bs + (wN * 32 + nt * 16) * GLDA, GLDA, lane);
#pragma unroll
    for (int mt = 0; mt < 4; ++mt)
#pragma unroll
      for (int nt = 0; nt < 2; ++nt)
        acc[mt][nt] = wmma_bf16(afr[mt], bfr[nt], acc[mt][nt]);
  }

  // ---- epilogue: C layout = row r+(lane>=16?8:0), col lane&15 ----
  const int rowHalf = (lane < 16) ? 0 : 8;
#pragma unroll
  for (int mt = 0; mt < 4; ++mt) {
#pragma unroll
    for (int nt = 0; nt < 2; ++nt) {
      int gc = n0 + wN * 32 + nt * 16 + (lane & 15);
#pragma unroll
      for (int r = 0; r < 8; ++r) {
        int gr  = m0 + wM * 64 + mt * 16 + r + rowHalf;
        float v = acc[mt][nt][r];
        if (OUT_F32) {
          ((float*)Cout)[(size_t)gr * N + gc] = v + bias[gc];
        } else {
          ((__bf16*)Cout)[(size_t)gr * N + gc] = f32_to_bf16(v);
        }
      }
    }
  }
}

// ---------------------------------------------------------------------------
// Flash attention: grid = (NQ/64, B*H), 128 threads (4 waves x 16 Q rows).
// K tiles staged by the Tensor Data Mover (with LDS pad to the 72-elem row
// stride); V staged transposed by VALU; online softmax in f32 (exp2 domain,
// scale*log2e folded into Q); QK^T and PV both on WMMA.
// ---------------------------------------------------------------------------
#define ALD 72    // padded LDS row stride: 144B rows, 16B aligned

__device__ __forceinline__ void tdm_load_tile_64x64_bf16(const void* gsrc,
                                                         void* ldsdst,
                                                         unsigned row_stride_elems) {
  unsigned long long ga = (unsigned long long)(size_t)gsrc;
  v4u g0;
  g0.x = 1u;                                        // count=1, user mode
  g0.y = (unsigned)(size_t)ldsdst;                  // lds_addr (bytes)
  g0.z = (unsigned)ga;                              // global_addr[31:0]
  g0.w = (unsigned)((ga >> 32) & 0x01FFFFFFull) | (2u << 30);  // [56:32] | type=2
  v8i g1;
  // data_size=1(2B) | pad_enable | pad_interval=4(32 dwords) | pad_amount=3(4 dwords)
  g1[0] = (1 << 16) | (1 << 20) | (4 << 22) | (3 << 25);
  g1[1] = 64 << 16;                                 // tensor_dim0 = 64 elems
  g1[2] = 64 << 16;                                 // tensor_dim1 = 64 rows
  g1[3] = 64 << 16;                                 // tile_dim0 = 64
  g1[4] = 64;                                       // tile_dim1 = 64 (tile_dim2=0)
  g1[5] = (int)row_stride_elems;                    // tensor_dim0_stride
  g1[6] = 0;
  g1[7] = 0;
  v4i z4 = {0, 0, 0, 0};
#if defined(__clang_major__) && (__clang_major__ >= 23)
  v8i z8 = {0, 0, 0, 0, 0, 0, 0, 0};
  __builtin_amdgcn_tensor_load_to_lds(g0, g1, z4, z4, z8, 0);
#else
  __builtin_amdgcn_tensor_load_to_lds(g0, g1, z4, z4, 0);
#endif
}

__global__ __launch_bounds__(128) void attn_kernel(const __bf16* __restrict__ Qp,
                                                   const __bf16* __restrict__ Kp,
                                                   const __bf16* __restrict__ Vp,
                                                   __bf16* __restrict__ X) {
  __shared__ __bf16 Qs[64 * ALD];
  __shared__ __bf16 Ks[64 * ALD];
  __shared__ __bf16 Vt[64 * ALD];   // V transposed: Vt[d][k]
  __shared__ __bf16 Ps[64 * ALD];   // per-wave P staging (C-layout -> A-layout)

  const int qb   = blockIdx.x;            // 64-row Q block
  const int bh   = blockIdx.y;
  const int b    = bh >> 4;
  const int h    = bh & 15;
  const int tid  = threadIdx.x;
  const int lane = tid & 31;
  const int wv   = tid >> 5;
  const int r0   = wv * 16;               // this wave's Q rows within block

  const size_t qrow0 = (size_t)b * 2048 + qb * 64;
  const size_t krow0 = (size_t)b * 2048;
  const int    hc    = h * 64;
  // fold softmax scale and log2(e) into Q -> P = exp2(S - m)
  const float qscale = 0.125f * 1.44269504088896340736f;

  // load + scale Q block (64 x 64)
#pragma unroll
  for (int i = 0; i < 32; ++i) {
    int idx = tid + i * 128;
    int d = idx & 63, q = idx >> 6;
    float v = bf16_to_f32(Qp[(qrow0 + q) * 1024 + hc + d]) * qscale;
    Qs[q * ALD + d] = f32_to_bf16(v);
  }

  v8f   o[4];
  float m_r[8], l_r[8];
#pragma unroll
  for (int nt = 0; nt < 4; ++nt) o[nt] = v8f_zero();
#pragma unroll
  for (int r = 0; r < 8; ++r) { m_r[r] = -1e30f; l_r[r] = 0.0f; }

  for (int kb = 0; kb < 2048; kb += 64) {
    __syncthreads();
    // stage K tile (64 x 64, row-major, padded rows) via Tensor Data Mover
    if (wv == 0) {
      tdm_load_tile_64x64_bf16(Kp + (krow0 + kb) * 1024 + hc, Ks, 1024u);
    }
    // stage V transposed: Vt[d][k] = V[kb+k][hc+d] (coalesced over d)
#pragma unroll
    for (int i = 0; i < 32; ++i) {
      int idx = tid + i * 128;
      int d = idx & 63, k = idx >> 6;
      Vt[d * ALD + k] = Vp[(krow0 + kb + k) * 1024 + hc + d];
    }
    if (wv == 0) __builtin_amdgcn_s_wait_tensorcnt(0);
    __syncthreads();

    // ---- S = Q K^T (16x64 per wave) ----
    v8f s[4];
#pragma unroll
    for (int nt = 0; nt < 4; ++nt) s[nt] = v8f_zero();
#pragma unroll
    for (int ks = 0; ks < 64; ks += 32) {
      v16bf aq = load_a_frag(Qs + r0 * ALD + ks, ALD, lane);
#pragma unroll
      for (int nt = 0; nt < 4; ++nt) {
        v16bf bk = load_b_frag(Ks + (nt * 16) * ALD + ks, ALD, lane);
        s[nt] = wmma_bf16(aq, bk, s[nt]);
      }
    }

    // ---- online softmax (per C-fragment row; rows live across 16 lanes) ----
#pragma unroll
    for (int r = 0; r < 8; ++r) {
      float mx = fmaxf(fmaxf(s[0][r], s[1][r]), fmaxf(s[2][r], s[3][r]));
#pragma unroll
      for (int msk = 8; msk >= 1; msk >>= 1) mx = fmaxf(mx, __shfl_xor(mx, msk, 32));
      float mnew  = fmaxf(m_r[r], mx);
      float alpha = exp2f(m_r[r] - mnew);
      m_r[r] = mnew;
      float rs = 0.0f;
#pragma unroll
      for (int nt = 0; nt < 4; ++nt) {
        float p = exp2f(s[nt][r] - mnew);
        s[nt][r] = p;
        rs += p;
      }
#pragma unroll
      for (int msk = 8; msk >= 1; msk >>= 1) rs += __shfl_xor(rs, msk, 32);
      l_r[r] = l_r[r] * alpha + rs;
#pragma unroll
      for (int nt = 0; nt < 4; ++nt) o[nt][r] = o[nt][r] * alpha;
    }

    // ---- restage P: C-layout f32 -> bf16 A-layout via per-wave LDS ----
    {
      const int rowHalf = (lane < 16) ? 0 : 8;
      const int col     = lane & 15;
#pragma unroll
      for (int nt = 0; nt < 4; ++nt)
#pragma unroll
        for (int r = 0; r < 8; ++r)
          Ps[(r0 + r + rowHalf) * ALD + nt * 16 + col] = f32_to_bf16(s[nt][r]);
    }

    // ---- O += P V ----
#pragma unroll
    for (int ks = 0; ks < 64; ks += 32) {
      v16bf ap = load_a_frag(Ps + r0 * ALD + ks, ALD, lane);
#pragma unroll
      for (int nt = 0; nt < 4; ++nt) {
        v16bf bv = load_b_frag(Vt + (nt * 16) * ALD + ks, ALD, lane);
        o[nt] = wmma_bf16(ap, bv, o[nt]);
      }
    }
  }

  // ---- normalize and write X (bf16, heads re-interleaved) ----
  const int rowHalf = (lane < 16) ? 0 : 8;
#pragma unroll
  for (int r = 0; r < 8; ++r) {
    float inv = 1.0f / l_r[r];
    int row = r0 + r + rowHalf;
#pragma unroll
    for (int nt = 0; nt < 4; ++nt) {
      int col = nt * 16 + (lane & 15);
      X[(qrow0 + row) * 1024 + hc + col] = f32_to_bf16(o[nt][r] * inv);
    }
  }
}

// ---------------------------------------------------------------------------
// Launch: 3 projection GEMMs -> flash attention -> output GEMM (+bias).
// Workspace: Qp,Kp,Vp,X bf16 (4 x 8 MB = 32 MB).
// ---------------------------------------------------------------------------
extern "C" void kernel_launch(void* const* d_in, const int* in_sizes, int n_in,
                              void* d_out, int out_size, void* d_ws, size_t ws_size,
                              hipStream_t stream) {
  const float* query = (const float*)d_in[0];
  const float* keyi  = (const float*)d_in[1];
  const float* value = (const float*)d_in[2];
  // d_in[3]=qpos, d_in[4]=kpos unused (rope is None in reference)
  const float* Wq = (const float*)d_in[5];
  const float* Wk = (const float*)d_in[6];
  const float* Wv = (const float*)d_in[7];
  const float* Wo = (const float*)d_in[8];
  const float* bo = (const float*)d_in[9];

  const int Bsz = 2, NQ = 2048, DIM = 1024, HEADS = 16;
  const int M = Bsz * NQ;                       // 4096
  const size_t nElem = (size_t)M * DIM;

  __bf16* Qp = (__bf16*)d_ws;
  __bf16* Kp = Qp + nElem;
  __bf16* Vp = Kp + nElem;
  __bf16* Xa = Vp + nElem;
  (void)ws_size; (void)in_sizes; (void)n_in; (void)out_size;

  dim3 gg(DIM / 128, M / 128);                  // (8, 32)
  gemm_kernel<true, false><<<gg, 256, 0, stream>>>(query, Wq, Qp, nullptr, M, DIM, DIM);
  gemm_kernel<true, false><<<gg, 256, 0, stream>>>(keyi,  Wk, Kp, nullptr, M, DIM, DIM);
  gemm_kernel<true, false><<<gg, 256, 0, stream>>>(value, Wv, Vp, nullptr, M, DIM, DIM);

  dim3 ga(NQ / 64, Bsz * HEADS);                // (32, 32)
  attn_kernel<<<ga, 128, 0, stream>>>(Qp, Kp, Vp, Xa);

  gemm_kernel<false, true><<<gg, 256, 0, stream>>>(Xa, Wo, (float*)d_out, bo, M, DIM, DIM);
}
